// AttentionMemory_16930761081022
// MI455X (gfx1250) — matrix operand order repaired and checked by reading, not compiled, not run on hardware
//
#include <hip/hip_runtime.h>

// Problem constants (reference: B=8, CK=64, CV=512, H=W=64)
constexpr int kBat = 8;
constexpr int kCK  = 64;
constexpr int kCV  = 512;
constexpr int kHW  = 64 * 64;

constexpr int QT  = 64;    // queries per block
constexpr int MT  = 64;    // memory (key) chunk per iteration
constexpr int CVT = 256;   // value channels per block (halves GEMM1 redundancy)

typedef __bf16 bf16_t;
typedef __attribute__((ext_vector_type(16))) __bf16 v16bf;
typedef __attribute__((ext_vector_type(8)))  __bf16 v8bf;
typedef __attribute__((ext_vector_type(8)))  float  v8f;

static __device__ __forceinline__ unsigned short f32_to_bf16_bits(float f) {
  // Native hardware convert (v_cvt path) instead of 3-op manual RNE.
  return __builtin_bit_cast(unsigned short, static_cast<__bf16>(f));
}

// Two aligned 16B LDS loads -> one 16-element bf16 fragment.
static __device__ __forceinline__ v16bf ld_frag(const unsigned short* p0,
                                                const unsigned short* p1) {
  v8bf lo = *reinterpret_cast<const v8bf*>(p0);
  v8bf hi = *reinterpret_cast<const v8bf*>(p1);
  return __builtin_shufflevector(lo, hi, 0, 1, 2, 3, 4, 5, 6, 7,
                                         8, 9, 10, 11, 12, 13, 14, 15);
}

static __device__ __forceinline__ unsigned long long pack4_bf16(float a, float b,
                                                                float c, float d) {
  return (unsigned long long)f32_to_bf16_bits(a)
       | ((unsigned long long)f32_to_bf16_bits(b) << 16)
       | ((unsigned long long)f32_to_bf16_bits(c) << 32)
       | ((unsigned long long)f32_to_bf16_bits(d) << 48);
}

__global__ __launch_bounds__(256) void
flash_attention_mem_kernel(const float* __restrict__ Mk,
                           const float* __restrict__ Qk,
                           const float* __restrict__ mv,
                           float* __restrict__ out) {
  // LDS layouts: every WMMA fragment element run contiguous per lane.
  __shared__ __align__(16) unsigned short Qs[QT][kCK + 8];   // Q^T  [q][k] (scaled)
  __shared__ __align__(16) unsigned short Ms[MT][kCK + 8];   // Mk^T [m][k]
  __shared__ __align__(16) unsigned short Ps[QT][MT + 8];    // P^T  [q][m]
  __shared__ __align__(16) unsigned short Vs[CVT][MT + 8];   // V    [cv][m]
  __shared__ __align__(16) float Ss[QT][MT + 4];             // S^T  [q][m]
  __shared__ float run_max[QT];
  __shared__ float run_sum[QT];
  __shared__ float row_scale[QT];
  __shared__ float pmax[4][QT];   // per-segment partial max
  __shared__ float psum[4][QT];   // per-segment partial sum

  const int b    = blockIdx.z;
  const int q0   = blockIdx.x * QT;
  const int cv0  = blockIdx.y * CVT;
  const int tid  = threadIdx.x;
  const int lane = tid & 31;
  const int wave = tid >> 5;

  // WMMA lane decomposition (ISA 7.12.2, wave32)
  const int n    = lane & 15;               // column for B/C/D, row for A
  const int kA   = (lane < 16) ? 0 : 8;     // A-fragment K base
  const int kB16 = (lane < 16) ? 0 : 16;    // unused (B fragments also use kA)
  const int mb   = (lane < 16) ? 0 : 8;     // C/D M base

  const float scale = 0.125f;               // 1/sqrt(CK)

  const float* Qb = Qk + (size_t)b * kCK * kHW;
  const float* Mb = Mk + (size_t)b * kCK * kHW;
  const float* Vb = mv + (size_t)b * kCV * kHW + (size_t)cv0 * kHW;

  // 4x4 transpose-staging coordinates.
  const int k4 = tid >> 4;   // 0..15 (k quad)
  const int x4 = tid & 15;   // 0..15 (m/q quad)

  // Softmax parallelization coordinates: 4 segments x 64 queries.
  const int sq  = tid & 63;  // query
  const int seg = tid >> 6;  // m-segment (16 wide)

  // ---- Stage Q^T once: coalesced float4 reads, register 4x4 transpose ----
  {
    float4 f[4];
    #pragma unroll
    for (int r = 0; r < 4; ++r)
      f[r] = *reinterpret_cast<const float4*>(Qb + (size_t)(k4 * 4 + r) * kHW +
                                              q0 + x4 * 4);
    const float* fp = reinterpret_cast<const float*>(f);
    #pragma unroll
    for (int j = 0; j < 4; ++j) {
      *reinterpret_cast<unsigned long long*>(&Qs[x4 * 4 + j][k4 * 4]) =
          pack4_bf16(fp[0 * 4 + j] * scale, fp[1 * 4 + j] * scale,
                     fp[2 * 4 + j] * scale, fp[3 * 4 + j] * scale);
    }
  }
  if (tid < QT) {
    run_max[tid] = -3.0e38f;
    run_sum[tid] = 0.0f;
  }

  // Accumulators: wave owns cv rows [32*wave, 32*wave+32) as 2 tiles, 4 q-tiles.
  v8f acc[2][4];
  #pragma unroll
  for (int c = 0; c < 2; ++c)
    #pragma unroll
    for (int t = 0; t < 4; ++t)
      #pragma unroll
      for (int e = 0; e < 8; ++e) acc[c][t][e] = 0.0f;

  __syncthreads();

  for (int m0 = 0; m0 < kHW; m0 += MT) {
    // Prefetch next chunk while computing this one (global_prefetch_b8).
    if (m0 + MT < kHW) {
      __builtin_prefetch(Mb + (size_t)(k4 * 4) * kHW + (m0 + MT) + x4 * 4, 0, 3);
      __builtin_prefetch(Vb + (size_t)tid * kHW + (m0 + MT), 0, 3);
    }

    // ---- Stage Mk^T [m][k]: float4 reads + register 4x4 transpose ----
    {
      float4 f[4];
      #pragma unroll
      for (int r = 0; r < 4; ++r)
        f[r] = *reinterpret_cast<const float4*>(Mb + (size_t)(k4 * 4 + r) * kHW +
                                                m0 + x4 * 4);
      const float* fp = reinterpret_cast<const float*>(f);
      #pragma unroll
      for (int j = 0; j < 4; ++j) {
        *reinterpret_cast<unsigned long long*>(&Ms[x4 * 4 + j][k4 * 4]) =
            pack4_bf16(fp[0 * 4 + j], fp[1 * 4 + j], fp[2 * 4 + j], fp[3 * 4 + j]);
      }
    }
    // ---- Stage V [cv][m]: packed b64 stores, 16 float4 per thread ----
    #pragma unroll 4
    for (int i = tid; i < CVT * MT / 4; i += 256) {
      int c   = i >> 4;
      int m4i = (i & 15) << 2;
      float4 v = *reinterpret_cast<const float4*>(Vb + (size_t)c * kHW + m0 + m4i);
      *reinterpret_cast<unsigned long long*>(&Vs[c][m4i]) =
          pack4_bf16(v.x, v.y, v.z, v.w);
    }
    __syncthreads();

    // ---- GEMM1: S^T[q][m] = sum_k Mk^T[m][k] * Q^T[q][k] ----
    {
      const int tm   = wave >> 1;
      const int tq0  = (wave & 1) << 1;
      const int mrow = tm * 16 + n;
      v8f s0 = {0.f, 0.f, 0.f, 0.f, 0.f, 0.f, 0.f, 0.f};
      v8f s1 = {0.f, 0.f, 0.f, 0.f, 0.f, 0.f, 0.f, 0.f};
      #pragma unroll
      for (int kk = 0; kk < kCK; kk += 32) {
        v16bf a  = ld_frag(&Ms[mrow][kk + kA], &Ms[mrow][kk + kA + 16]);
        v16bf b0 = ld_frag(&Qs[tq0 * 16 + n][kk + kA],
                           &Qs[tq0 * 16 + n][kk + kA + 16]);
        v16bf b1 = ld_frag(&Qs[(tq0 + 1) * 16 + n][kk + kA],
                           &Qs[(tq0 + 1) * 16 + n][kk + kA + 16]);
        s0 = __builtin_amdgcn_wmma_f32_16x16x32_bf16(false, a, false, b0,
                                                     (short)0, s0, false, false);
        s1 = __builtin_amdgcn_wmma_f32_16x16x32_bf16(false, a, false, b1,
                                                     (short)0, s1, false, false);
      }
      float* d0 = &Ss[tq0 * 16 + n][tm * 16 + mb];
      float* d1 = &Ss[(tq0 + 1) * 16 + n][tm * 16 + mb];
      float4 w0 = {s0[0], s0[1], s0[2], s0[3]}, w1 = {s0[4], s0[5], s0[6], s0[7]};
      float4 w2 = {s1[0], s1[1], s1[2], s1[3]}, w3 = {s1[4], s1[5], s1[6], s1[7]};
      *reinterpret_cast<float4*>(d0)     = w0;
      *reinterpret_cast<float4*>(d0 + 4) = w1;
      *reinterpret_cast<float4*>(d1)     = w2;
      *reinterpret_cast<float4*>(d1 + 4) = w3;
    }
    __syncthreads();

    // ---- Online softmax, parallel over all 256 threads ----
    // Stage A: partial max per (segment, q).
    {
      const float4* srow = reinterpret_cast<const float4*>(&Ss[sq][seg * 16]);
      float cm = -3.0e38f;
      #pragma unroll
      for (int i = 0; i < 4; ++i) {
        float4 v = srow[i];
        cm = fmaxf(cm, fmaxf(fmaxf(v.x, v.y), fmaxf(v.z, v.w)));
      }
      pmax[seg][sq] = cm;
    }
    __syncthreads();
    // Stage B: combine maxes, compute rescale factor.
    if (tid < QT) {
      const int q = tid;
      const float omx = run_max[q];
      float cm = fmaxf(fmaxf(pmax[0][q], pmax[1][q]),
                       fmaxf(pmax[2][q], pmax[3][q]));
      const float nmx = fmaxf(omx, cm);
      row_scale[q] = __expf(omx - nmx);
      run_max[q]   = nmx;
    }
    __syncthreads();
    // Stage C: parallel exp, write P^T, partial sums.
    {
      const float nmx = run_max[sq];
      const float4* srow = reinterpret_cast<const float4*>(&Ss[sq][seg * 16]);
      float ssum = 0.0f;
      #pragma unroll
      for (int i = 0; i < 4; ++i) {
        float4 v = srow[i];
        float e0 = __expf(v.x - nmx), e1 = __expf(v.y - nmx);
        float e2 = __expf(v.z - nmx), e3 = __expf(v.w - nmx);
        ssum += (e0 + e1) + (e2 + e3);
        *reinterpret_cast<unsigned long long*>(&Ps[sq][seg * 16 + i * 4]) =
            pack4_bf16(e0, e1, e2, e3);
      }
      psum[seg][sq] = ssum;
    }
    __syncthreads();
    // Stage D: fold partial sums into running denominator (no one reads
    // run_sum until the epilogue, so no extra barrier needed).
    if (tid < QT) {
      const int q = tid;
      run_sum[q] = run_sum[q] * row_scale[q] +
                   ((psum[0][q] + psum[1][q]) + (psum[2][q] + psum[3][q]));
    }

    // ---- Rescale accumulators; GEMM2: O[cv][q] += V[cv][m] * P^T[q][m] ----
    #pragma unroll
    for (int tq = 0; tq < 4; ++tq) {
      const float alpha = row_scale[tq * 16 + n];
      #pragma unroll
      for (int c = 0; c < 2; ++c)
        #pragma unroll
        for (int v = 0; v < 8; ++v) acc[c][tq][v] *= alpha;
    }
    {
      const int crow0 = wave * 32 + n;
      const int crow1 = wave * 32 + 16 + n;
      #pragma unroll
      for (int kk = 0; kk < MT; kk += 32) {
        v16bf a0 = ld_frag(&Vs[crow0][kk + kA], &Vs[crow0][kk + kA + 16]);
        v16bf a1 = ld_frag(&Vs[crow1][kk + kA], &Vs[crow1][kk + kA + 16]);
        #pragma unroll
        for (int tq = 0; tq < 4; ++tq) {
          v16bf bfr = ld_frag(&Ps[tq * 16 + n][kk + kA],
                              &Ps[tq * 16 + n][kk + kA + 16]);
          acc[0][tq] = __builtin_amdgcn_wmma_f32_16x16x32_bf16(
              false, a0, false, bfr, (short)0, acc[0][tq], false, false);
          acc[1][tq] = __builtin_amdgcn_wmma_f32_16x16x32_bf16(
              false, a1, false, bfr, (short)0, acc[1][tq], false, false);
        }
      }
    }
    __syncthreads();
  }

  // ---- Epilogue: divide by softmax sum, store fp32 output [b][cv][q] ----
  float* Ob = out + (size_t)b * kCV * kHW;
  #pragma unroll
  for (int tq = 0; tq < 4; ++tq) {
    const float inv = 1.0f / run_sum[tq * 16 + n];
    const int q = q0 + tq * 16 + n;
    #pragma unroll
    for (int c = 0; c < 2; ++c) {
      #pragma unroll
      for (int v = 0; v < 8; ++v) {
        const int cv = cv0 + wave * 32 + c * 16 + mb + v;
        Ob[(size_t)cv * kHW + q] = acc[c][tq][v] * inv;
      }
    }
  }
}

extern "C" void kernel_launch(void* const* d_in, const int* in_sizes, int n_in,
                              void* d_out, int out_size, void* d_ws, size_t ws_size,
                              hipStream_t stream) {
  const float* Mk = (const float*)d_in[0];
  const float* Qk = (const float*)d_in[1];
  const float* mv = (const float*)d_in[2];
  float* out = (float*)d_out;
  (void)in_sizes; (void)n_in; (void)out_size; (void)d_ws; (void)ws_size;

  dim3 grid(kHW / QT, kCV / CVT, kBat);   // 64 x 2 x 8 = 1024 blocks
  flash_attention_mem_kernel<<<grid, dim3(256, 1, 1), 0, stream>>>(Mk, Qk, mv, out);
}
